// SwinAttentionBlock_27255862460629
// MI455X (gfx1250) — matrix operand order repaired and checked by reading, not compiled, not run on hardware
//
#include <hip/hip_runtime.h>
#include <hip/hip_bf16.h>
#include <cstdint>

typedef _Float16 v16h __attribute__((ext_vector_type(16)));
typedef _Float16 h8   __attribute__((ext_vector_type(8)));
typedef float    v8f  __attribute__((ext_vector_type(8)));
typedef float    v4f  __attribute__((ext_vector_type(4)));
typedef unsigned int u32x4 __attribute__((ext_vector_type(4)));
typedef int      i32x4v __attribute__((ext_vector_type(4)));
typedef int      i32x8v __attribute__((ext_vector_type(8)));

#if defined(__gfx1250__) && __has_builtin(__builtin_amdgcn_tensor_load_to_lds)
#define USE_TDM 1
#else
#define USE_TDM 0
#endif

#if USE_TDM
// raw LDS byte offset of a __shared__ object
__device__ __forceinline__ unsigned lds_addr_of(const void* p) {
  return (unsigned)(unsigned long long)(__attribute__((address_space(3))) const void*)p;
}
// 2D TDM: tile 32 halfs x 128 rows, row stride = stride_halfs (f16), into LDS with
// 16B pad after every 64B row -> LDS row stride 40 halfs (matches As/Bs layout).
__device__ __forceinline__ void tdm_load_tile(unsigned lds, const _Float16* g,
                                              int stride_halfs) {
  unsigned long long ga = (unsigned long long)(uintptr_t)g;
  u32x4 g0;
  g0.x = 1u;                                  // count=1, user mode, no gather
  g0.y = lds;                                 // lds_addr
  g0.z = (unsigned)ga;                        // global_addr[31:0]
  g0.w = (unsigned)(ga >> 32) | (2u << 30);   // global_addr[56:32] | type=2
  i32x8v g1;
  g1[0] = (1 << 16) | (1 << 20) | (3 << 22) | (3 << 25); // data_size=2B, pad_en, 16dw int, 4dw amt
  g1[1] = (int)(0x4000u << 16);               // tensor_dim0 = 0x4000 (no clipping)
  g1[2] = 0;                                  // tensor_dim0 hi, tensor_dim1 lo
  g1[3] = (int)((32u << 16) | 0x10u);         // tensor_dim1 = 1M ; tile_dim0 = 32
  g1[4] = 128;                                // tile_dim1 = 128, tile_dim2 = 0
  g1[5] = stride_halfs;                       // tensor_dim0_stride (data_size units)
  g1[6] = 0;
  g1[7] = 0;
  i32x4v z4 = {0, 0, 0, 0};
  i32x8v z8 = {0, 0, 0, 0, 0, 0, 0, 0};
  __builtin_amdgcn_tensor_load_to_lds(g0, g1, z4, z4, z8, 0);
}
#endif

// ---------------- small helpers ----------------
__global__ __launch_bounds__(256) void cvt_f32_to_f16(const float* __restrict__ s,
                                                      _Float16* __restrict__ d, int n) {
  int i = blockIdx.x * 256 + threadIdx.x;
  if (i < n) d[i] = (_Float16)s[i];
}

__global__ __launch_bounds__(256) void zero_h8(h8* __restrict__ d) {
  h8 z;
#pragma unroll
  for (int e = 0; e < 8; e++) z[e] = (_Float16)0.f;
  d[(size_t)blockIdx.x * 256 + threadIdx.x] = z;
}

// bias+mask table, transposed & padded: bmt[wq][head][col 64][row 64]
__global__ __launch_bounds__(256) void build_bmt(const float* __restrict__ rpb,
                                                 float* __restrict__ bmt) {
  const int wq = blockIdx.x / 12, head = blockIdx.x % 12;
  const int wh = wq / 8, ww = wq % 8;
  for (int e = threadIdx.x; e < 64 * 64; e += 256) {
    int col = e >> 6, row = e & 63;
    float val = -1e30f;
    if (row < 49 && col < 49) {
      int i1 = row / 7, j1 = row % 7, i2 = col / 7, j2 = col % 7;
      int idx = (i1 - i2 + 6) * 13 + (j1 - j2 + 6);
      float bias = rpb[idx * 12 + head];
      auto cat = [](int u) { return u < 49 ? 0 : (u < 53 ? 1 : 2); };
      int r1 = 3 * cat(wh * 7 + i1) + cat(ww * 7 + j1);
      int r2 = 3 * cat(wh * 7 + i2) + cat(ww * 7 + j2);
      val = bias + ((r1 != r2) ? -100.0f : 0.0f);
    }
    bmt[(size_t)blockIdx.x * 4096 + e] = val;
  }
}

// ---------------- LN1 + shift + window gather ----------------
__global__ __launch_bounds__(256) void ln1_window(const float* __restrict__ x,
                                                  const float* __restrict__ gw,
                                                  const float* __restrict__ gb,
                                                  _Float16* __restrict__ xln) {
  const int wave = threadIdx.x >> 5, lane = threadIdx.x & 31;
  const int t = blockIdx.x * 8 + wave;
  const int bi = t / 3136, pos = t % 3136;
  const int h = pos / 56, wcol = pos % 56;
  float v[12];
  float s = 0.f;
#pragma unroll
  for (int i = 0; i < 12; i++) {
    int c = lane + 32 * i;
    v[i] = x[((size_t)(bi * 384 + c)) * 3136 + pos];
    s += v[i];
  }
#pragma unroll
  for (int off = 1; off < 32; off <<= 1) s += __shfl_xor(s, off, 32);
  float mean = s * (1.f / 384.f);
  float var = 0.f;
#pragma unroll
  for (int i = 0; i < 12; i++) { float d = v[i] - mean; var += d * d; }
#pragma unroll
  for (int off = 1; off < 32; off <<= 1) var += __shfl_xor(var, off, 32);
  float rstd = rsqrtf(var * (1.f / 384.f) + 1e-5f);
  int hs = (h + 53) % 56, ws2 = (wcol + 53) % 56;
  int wh = hs / 7, ii = hs % 7, ww = ws2 / 7, jj = ws2 % 7;
  int widx = bi * 64 + wh * 8 + ww;
  int n = ii * 7 + jj;
  size_t dst = ((size_t)(widx * 49 + n)) * 384;
#pragma unroll
  for (int i = 0; i < 12; i++) {
    int c = lane + 32 * i;
    xln[dst + c] = (_Float16)((v[i] - mean) * rstd * gw[c] + gb[c]);
  }
}

// ---------------- xres = shortcut + proj ; LN2 -> hin ----------------
__global__ __launch_bounds__(256) void xres_ln2(const float* __restrict__ x,
                                                const float* __restrict__ proj,
                                                const float* __restrict__ gw,
                                                const float* __restrict__ gb,
                                                float* __restrict__ xres,
                                                _Float16* __restrict__ hin) {
  const int wave = threadIdx.x >> 5, lane = threadIdx.x & 31;
  const int wt = blockIdx.x * 8 + wave;
  const int win = wt / 49, n = wt % 49;
  const int bi = win / 64, wq = win % 64;
  const int wh = wq / 8, ww = wq % 8;
  const int ii = n / 7, jj = n % 7;
  const int h = (wh * 7 + ii + 3) % 56, wcol = (ww * 7 + jj + 3) % 56;
  const int pos = h * 56 + wcol;
  const int t = bi * 3136 + pos;
  const size_t prow = ((size_t)(win * 64 + n)) * 384;
  float v[12];
  float s = 0.f;
#pragma unroll
  for (int i = 0; i < 12; i++) {
    int c = lane + 32 * i;
    v[i] = x[((size_t)(bi * 384 + c)) * 3136 + pos] + proj[prow + c];
    s += v[i];
  }
#pragma unroll
  for (int off = 1; off < 32; off <<= 1) s += __shfl_xor(s, off, 32);
  float mean = s * (1.f / 384.f);
  float var = 0.f;
#pragma unroll
  for (int i = 0; i < 12; i++) { float d = v[i] - mean; var += d * d; }
#pragma unroll
  for (int off = 1; off < 32; off <<= 1) var += __shfl_xor(var, off, 32);
  float rstd = rsqrtf(var * (1.f / 384.f) + 1e-5f);
#pragma unroll
  for (int i = 0; i < 12; i++) {
    int c = lane + 32 * i;
    xres[(size_t)t * 384 + c] = v[i];
    hin[(size_t)t * 384 + c] = (_Float16)((v[i] - mean) * rstd * gw[c] + gb[c]);
  }
}

// ---------------- generic WMMA GEMM: C = A(MxK,f16) * W(NxK,f16)^T + bias ----------------
// block tile 128x128, 8 waves, double-buffered LDS; tiles staged by TDM (tensor_load_to_lds)
// when available, else per-thread loads.
template <int EPI>
__global__ __launch_bounds__(256) void gemm_wmma(const _Float16* __restrict__ A,
                                                 const _Float16* __restrict__ W,
                                                 const float* __restrict__ bias,
                                                 _Float16* __restrict__ outH,
                                                 float* __restrict__ outF,
                                                 const float* __restrict__ xres,
                                                 _Float16* __restrict__ vT,
                                                 int M, int N, int K) {
  __shared__ _Float16 As[2][128][40];
  __shared__ _Float16 Bs[2][128][40];
  const int tid = threadIdx.x;
  const int wave = tid >> 5, lane = tid & 31;
  const int l16 = lane & 15, g = lane >> 4;
  const int mBase = blockIdx.y * 128;
  const int nBase = blockIdx.x * 128;
  const int mt = wave & 3;
  const int ntG = wave >> 2;
  const int nk = K >> 5;
  v8f acc[2][4];
#pragma unroll
  for (int su = 0; su < 2; su++)
#pragma unroll
    for (int i = 0; i < 4; i++) acc[su][i] = v8f{};

#if USE_TDM
  if (wave == 0) {
    tdm_load_tile(lds_addr_of(&As[0][0][0]), A + (size_t)mBase * K, K);
    tdm_load_tile(lds_addr_of(&Bs[0][0][0]), W + (size_t)nBase * K, K);
  }
#else
  const int ar = tid >> 1;          // 0..127
  const int ak = (tid & 1) * 16;    // 0 or 16
  {
    *(h8*)&As[0][ar][ak] = *(const h8*)&A[(size_t)(mBase + ar) * K + ak];
    *(h8*)&As[0][ar][ak + 8] = *(const h8*)&A[(size_t)(mBase + ar) * K + ak + 8];
    *(h8*)&Bs[0][ar][ak] = *(const h8*)&W[(size_t)(nBase + ar) * K + ak];
    *(h8*)&Bs[0][ar][ak + 8] = *(const h8*)&W[(size_t)(nBase + ar) * K + ak + 8];
  }
#endif

  for (int ks = 0; ks < nk; ks++) {
    const int cb = ks & 1;
#if USE_TDM
    __syncthreads();   // previous compute done -> other buffer free for overwrite
    if (wave == 0) {
      if (ks + 1 < nk) {
        const int k1 = (ks + 1) * 32;
        tdm_load_tile(lds_addr_of(&As[cb ^ 1][0][0]), A + (size_t)mBase * K + k1, K);
        tdm_load_tile(lds_addr_of(&Bs[cb ^ 1][0][0]), W + (size_t)nBase * K + k1, K);
        __builtin_amdgcn_s_wait_tensorcnt(2);   // in-order => stage ks complete
      } else {
        __builtin_amdgcn_s_wait_tensorcnt(0);
      }
    }
    __syncthreads();   // publish stage ks to all waves
#else
    __syncthreads();   // stage ks stores visible; previous compute done
    if (ks + 1 < nk) {
      const int k1 = (ks + 1) * 32;
      *(h8*)&As[cb ^ 1][ar][ak] = *(const h8*)&A[(size_t)(mBase + ar) * K + k1 + ak];
      *(h8*)&As[cb ^ 1][ar][ak + 8] = *(const h8*)&A[(size_t)(mBase + ar) * K + k1 + ak + 8];
      *(h8*)&Bs[cb ^ 1][ar][ak] = *(const h8*)&W[(size_t)(nBase + ar) * K + k1 + ak];
      *(h8*)&Bs[cb ^ 1][ar][ak + 8] = *(const h8*)&W[(size_t)(nBase + ar) * K + k1 + ak + 8];
    }
#endif
    v16h af[2];
#pragma unroll
    for (int su = 0; su < 2; su++) {
      const int row = mt * 32 + su * 16 + l16;
      h8 lo = *(const h8*)&As[cb][row][16 * g];
      h8 hi = *(const h8*)&As[cb][row][16 * g + 8];
#pragma unroll
      for (int e = 0; e < 8; e++) { af[su][e] = lo[e]; af[su][8 + e] = hi[e]; }
    }
#pragma unroll
    for (int nt = 0; nt < 4; nt++) {
      const int brow = ntG * 64 + nt * 16 + l16;
      h8 b0 = *(const h8*)&Bs[cb][brow][g * 16];
      h8 b1 = *(const h8*)&Bs[cb][brow][g * 16 + 8];
      v16h bf;
#pragma unroll
      for (int e = 0; e < 8; e++) { bf[e] = b0[e]; bf[8 + e] = b1[e]; }
#pragma unroll
      for (int su = 0; su < 2; su++)
        acc[su][nt] = __builtin_amdgcn_wmma_f32_16x16x32_f16(false, af[su], false, bf,
                                                             (short)0, acc[su][nt], false, false);
    }
  }
#pragma unroll
  for (int su = 0; su < 2; su++)
#pragma unroll
    for (int nt = 0; nt < 4; nt++) {
      const int col = nBase + ntG * 64 + nt * 16 + l16;
      const float bv = bias[col];
#pragma unroll
      for (int j = 0; j < 8; j++) {
        const int row = mBase + mt * 32 + su * 16 + j + 8 * g;
        float v = acc[su][nt][j] + bv;
        if (EPI == 0) {
          if (col < 768) {
            if (col < 384) v *= 0.17677669529663687f;      // 1/sqrt(32)
            outH[(size_t)row * 768 + col] = (_Float16)v;
          } else {
            const int dAll = col - 768;
            const int head = dAll >> 5, d = dAll & 31;
            const int win = row / 49, n = row % 49;
            vT[(((size_t)(win * 12 + head)) * 32 + d) * 64 + n] = (_Float16)v;
          }
        } else if (EPI == 1) {
          outF[(size_t)row * N + col] = v;
        } else if (EPI == 2) {
          float gv = 0.5f * v * (1.0f + erff(v * 0.70710678118654752f));
          outH[(size_t)row * N + col] = (_Float16)gv;
        } else {
          v += xres[(size_t)row * 384 + col];
          const int b = row / 3136, pos = row % 3136;
          outF[((size_t)(b * 384 + col)) * 3136 + pos] = v;
        }
      }
    }
}

// ---------------- windowed attention: one wave per (window, head) ----------------
__global__ __launch_bounds__(256) void attn_kernel(const _Float16* __restrict__ qk,
                                                   const _Float16* __restrict__ vT,
                                                   const float* __restrict__ bmt,
                                                   _Float16* __restrict__ att) {
  __shared__ _Float16 P[8][16][72];
  const int wave = threadIdx.x >> 5, lane = threadIdx.x & 31;
  const int l16 = lane & 15, g = lane >> 4;
  const int gid = blockIdx.x * 8 + wave;
  const int win = gid / 12, head = gid % 12;
  const int wq = win & 63;
  const size_t base = (size_t)win * 49 * 768;
  const int qoff = head * 32;
  const int koff = 384 + head * 32;
  const float* bmh = bmt + ((size_t)(wq * 12 + head)) * 4096;
  const _Float16* vbase = vT + (((size_t)(win * 12 + head)) * 32) * 64;

  v16h ones;
#pragma unroll
  for (int e = 0; e < 16; e++) ones[e] = (_Float16)1.f;

  v16h BK[4];
#pragma unroll
  for (int nt = 0; nt < 4; nt++) {
    int n = nt * 16 + l16;
    v16h b;
    if (n < 49) {
      const _Float16* p = qk + base + (size_t)n * 768 + koff + g * 16;
      h8 b0 = *(const h8*)p;
      h8 b1 = *(const h8*)(p + 8);
#pragma unroll
      for (int e = 0; e < 8; e++) { b[e] = b0[e]; b[8 + e] = b1[e]; }
    } else {
#pragma unroll
      for (int e = 0; e < 16; e++) b[e] = (_Float16)0.f;
    }
    BK[nt] = b;
  }
  v16h VB[4];
#pragma unroll
  for (int kt = 0; kt < 2; kt++)
#pragma unroll
    for (int nt2 = 0; nt2 < 2; nt2++) {
      const int d = nt2 * 16 + l16;
      const _Float16* p = vbase + (size_t)d * 64 + kt * 32 + g * 16;
      h8 b0 = *(const h8*)p;
      h8 b1 = *(const h8*)(p + 8);
      v16h b;
#pragma unroll
      for (int e = 0; e < 8; e++) { b[e] = b0[e]; b[8 + e] = b1[e]; }
      VB[kt * 2 + nt2] = b;
    }

  for (int mtile = 0; mtile < 4; mtile++) {
    v16h aq;
    {
      int m = mtile * 16 + l16;
      if (m < 49) {
        const _Float16* p = qk + base + (size_t)m * 768 + qoff;
        h8 lo = *(const h8*)(p + 16 * g);
        h8 hi = *(const h8*)(p + 16 * g + 8);
#pragma unroll
        for (int e = 0; e < 8; e++) { aq[e] = lo[e]; aq[8 + e] = hi[e]; }
      } else {
#pragma unroll
        for (int e = 0; e < 16; e++) aq[e] = (_Float16)0.f;
      }
    }
    v8f S[4];
#pragma unroll
    for (int nt = 0; nt < 4; nt++) {
      v8f z{};
      S[nt] = __builtin_amdgcn_wmma_f32_16x16x32_f16(false, aq, false, BK[nt],
                                                     (short)0, z, false, false);
    }
#pragma unroll
    for (int nt = 0; nt < 4; nt++) {
      const float* p = bmh + ((size_t)(nt * 16 + l16)) * 64 + mtile * 16 + 8 * g;
      v4f b0 = *(const v4f*)p;
      v4f b1 = *(const v4f*)(p + 4);
#pragma unroll
      for (int j = 0; j < 4; j++) S[nt][j] = __expf(S[nt][j] + b0[j]);
#pragma unroll
      for (int j = 0; j < 4; j++) S[nt][4 + j] = __expf(S[nt][4 + j] + b1[j]);
    }
#pragma unroll
    for (int nt = 0; nt < 4; nt++) {
      int col = nt * 16 + l16;
#pragma unroll
      for (int j = 0; j < 8; j++) P[wave][j + 8 * g][col] = (_Float16)S[nt][j];
    }
    v8f O[2], Os;
    O[0] = v8f{};
    O[1] = v8f{};
    Os = v8f{};
#pragma unroll
    for (int kt = 0; kt < 2; kt++) {
      v16h pa;
      const _Float16* pr = &P[wave][l16][0];
      h8 lo = *(const h8*)(pr + kt * 32 + 16 * g);
      h8 hi = *(const h8*)(pr + kt * 32 + 16 * g + 8);
#pragma unroll
      for (int e = 0; e < 8; e++) { pa[e] = lo[e]; pa[8 + e] = hi[e]; }
#pragma unroll
      for (int nt2 = 0; nt2 < 2; nt2++)
        O[nt2] = __builtin_amdgcn_wmma_f32_16x16x32_f16(false, pa, false, VB[kt * 2 + nt2],
                                                        (short)0, O[nt2], false, false);
      Os = __builtin_amdgcn_wmma_f32_16x16x32_f16(false, pa, false, ones,
                                                  (short)0, Os, false, false);
    }
#pragma unroll
    for (int nt2 = 0; nt2 < 2; nt2++) {
      int d = nt2 * 16 + l16;
#pragma unroll
      for (int j = 0; j < 8; j++) {
        int row = mtile * 16 + j + 8 * g;
        att[((size_t)(win * 64 + row)) * 384 + head * 32 + d] =
            (_Float16)(O[nt2][j] / (Os[j] + 1e-30f));
      }
    }
  }
}

// ---------------- host ----------------
extern "C" void kernel_launch(void* const* d_in, const int* in_sizes, int n_in,
                              void* d_out, int out_size, void* d_ws, size_t ws_size,
                              hipStream_t stream) {
  const float* x      = (const float*)d_in[0];
  const float* qkv_w  = (const float*)d_in[1];
  const float* qkv_b  = (const float*)d_in[2];
  const float* proj_w = (const float*)d_in[3];
  const float* proj_b = (const float*)d_in[4];
  const float* rpb    = (const float*)d_in[5];
  const float* n1w    = (const float*)d_in[6];
  const float* n1b    = (const float*)d_in[7];
  const float* n2w    = (const float*)d_in[8];
  const float* n2b    = (const float*)d_in[9];
  const float* fc1_w  = (const float*)d_in[10];
  const float* fc1_b  = (const float*)d_in[11];
  const float* fc2_w  = (const float*)d_in[12];
  const float* fc2_b  = (const float*)d_in[13];
  float* out = (float*)d_out;
  char* ws = (char*)d_ws;

  const int M = 50176;
  const int MP = 65536;

  size_t o = 0;
  _Float16* wqkvh = (_Float16*)(ws + o); o += (size_t)1152 * 384 * 2;
  _Float16* wprojh = (_Float16*)(ws + o); o += (size_t)384 * 384 * 2;
  _Float16* wfc1h = (_Float16*)(ws + o); o += (size_t)1536 * 384 * 2;
  _Float16* wfc2h = (_Float16*)(ws + o); o += (size_t)384 * 1536 * 2;
  float* bmt = (float*)(ws + o); o += (size_t)64 * 12 * 4096 * 4;
  char* RA = ws + o; o += (size_t)MP * 384 * 2;
  char* RB = ws + o; o += (size_t)M * 1536 * 2;
  char* RC = ws + o; o += (size_t)M * 384 * 4;
  (void)ws_size; (void)n_in; (void)in_sizes; (void)out_size;

  _Float16* xln    = (_Float16*)RA;
  _Float16* attout = (_Float16*)RA;
  _Float16* hin    = (_Float16*)RA;
  _Float16* qkh    = (_Float16*)RB;
  float*    projf  = (float*)RB;
  _Float16* fc1h   = (_Float16*)RB;
  _Float16* vT     = (_Float16*)RC;
  float*    xresf  = (float*)RC;

  cvt_f32_to_f16<<<(442368 + 255) / 256, 256, 0, stream>>>(qkv_w, wqkvh, 442368);
  cvt_f32_to_f16<<<(147456 + 255) / 256, 256, 0, stream>>>(proj_w, wprojh, 147456);
  cvt_f32_to_f16<<<(589824 + 255) / 256, 256, 0, stream>>>(fc1_w, wfc1h, 589824);
  cvt_f32_to_f16<<<(589824 + 255) / 256, 256, 0, stream>>>(fc2_w, wfc2h, 589824);
  zero_h8<<<(1024 * 12 * 32 * 64 / 8) / 256, 256, 0, stream>>>((h8*)vT);
  build_bmt<<<768, 256, 0, stream>>>(rpb, bmt);
  ln1_window<<<M / 8, 256, 0, stream>>>(x, n1w, n1b, xln);
  gemm_wmma<0><<<dim3(1152 / 128, M / 128), 256, 0, stream>>>(
      xln, wqkvh, qkv_b, qkh, nullptr, nullptr, vT, M, 1152, 384);
  attn_kernel<<<12288 / 8, 256, 0, stream>>>(qkh, vT, bmt, attout);
  gemm_wmma<1><<<dim3(384 / 128, MP / 128), 256, 0, stream>>>(
      attout, wprojh, proj_b, nullptr, projf, nullptr, nullptr, MP, 384, 384);
  xres_ln2<<<M / 8, 256, 0, stream>>>(x, projf, n2w, n2b, xresf, hin);
  gemm_wmma<2><<<dim3(1536 / 128, M / 128), 256, 0, stream>>>(
      hin, wfc1h, fc1_b, fc1h, nullptr, nullptr, nullptr, M, 1536, 384);
  gemm_wmma<3><<<dim3(384 / 128, M / 128), 256, 0, stream>>>(
      fc1h, wfc2h, fc2_b, nullptr, out, xresf, nullptr, M, 384, 1536);
}
